// LSTMDecoder_39444979647093
// MI455X (gfx1250) — matrix operand order repaired and checked by reading, not compiled, not run on hardware
//
#include <hip/hip_runtime.h>

// Problem dims (compile-time, match reference)
#define B_     256
#define T_     256
#define LAT_   128
#define H_     512
#define D_     128
#define G4H_   (4 * H_)

typedef __attribute__((ext_vector_type(16))) __bf16 v16bf;
typedef __attribute__((ext_vector_type(8)))  __bf16 v8bf;
typedef __attribute__((ext_vector_type(8)))  float  v8f;

union FragAB { v16bf v; v8bf h[2]; };

#define WMMA_BF16(A, B, C) \
  __builtin_amdgcn_wmma_f32_16x16x32_bf16(false, (A).v, false, (B).v, (short)0, (C), false, false)

// f32 -> bf16 round-to-nearest-even (bit twiddle; stored as u16)
__device__ __forceinline__ unsigned short f2bf(float x) {
  unsigned int u = __float_as_uint(x);
  u += 0x7FFFu + ((u >> 16) & 1u);
  return (unsigned short)(u >> 16);
}

__device__ __forceinline__ float sigmoidf_(float x) {
  return 1.0f / (1.0f + __expf(-x));
}
__device__ __forceinline__ float tanhf_(float x) {
  x = fminf(fmaxf(x, -15.0f), 15.0f);
  float e = __expf(-2.0f * x);
  return (1.0f - e) / (1.0f + e);
}

// ---------------- prep kernels ----------------

// WhT[n][k] = bf16(Wh[k][n]); Wh is (H, 4H) f32 row-major; WhT is (4H, H) bf16
__global__ void k_conv_wh(const float* __restrict__ Wh, unsigned short* __restrict__ WhT) {
  int idx = blockIdx.x * 256 + threadIdx.x;   // 0 .. 4H*H-1
  int k = idx & (H_ - 1);
  int n = idx >> 9;                            // H_ == 512
  WhT[idx] = f2bf(Wh[(size_t)k * G4H_ + n]);
}

// WoutT[t][d][k] = bf16(Wout[t][k][d]); Wout is (T,H,D) f32
__global__ void k_conv_wout(const float* __restrict__ Wout, unsigned short* __restrict__ WoutT) {
  size_t idx = (size_t)blockIdx.x * 256 + threadIdx.x;  // 0 .. T*D*H-1
  int k = (int)(idx & (H_ - 1));
  int d = (int)((idx >> 9) & (D_ - 1));
  int t = (int)(idx >> 16);                              // 9 + 7 bits
  WoutT[idx] = f2bf(Wout[(size_t)t * H_ * D_ + (size_t)k * D_ + d]);
}

// xg[n] = token @ Wi[:,n] + bh[n]  (fold bias in once)
__global__ void k_xgates(const float* __restrict__ token, const float* __restrict__ Wi,
                         const float* __restrict__ bh, float* __restrict__ xg) {
  for (int n = threadIdx.x; n < G4H_; n += 256) {
    float s = bh[n];
    #pragma unroll 4
    for (int k = 0; k < LAT_; ++k) s += token[k] * Wi[(size_t)k * G4H_ + n];
    xg[n] = s;
  }
}

// h0 = relu(z @ Wz + bz) -> bf16 buffer; c = 0
__global__ void k_h0(const float* __restrict__ z, const float* __restrict__ Wz,
                     const float* __restrict__ bz,
                     unsigned short* __restrict__ h0, float* __restrict__ c) {
  int idx = blockIdx.x * 256 + threadIdx.x;   // 0 .. B*H-1
  int b = idx >> 9;
  int j = idx & (H_ - 1);
  float s = bz[j];
  #pragma unroll 4
  for (int k = 0; k < LAT_; ++k) s += z[(size_t)b * LAT_ + k] * Wz[(size_t)k * H_ + j];
  s = fmaxf(s, 0.0f);
  h0[idx] = f2bf(s);
  c[idx] = 0.0f;
}

// ---------------- LSTM step: gates GEMM + cell update (fused) ----------------
// One block (128 threads = 4 waves) per 16x16 (batch x hidden) tile; wave g
// computes gate g's 16x16 GEMM tile (K=512, 16 WMMAs, depth-2 pipelined),
// then the 4 gate tiles meet in LDS for the fused c/h update.
// Grid: 512 blocks -> 2048 waves per step.
__global__ void k_lstm_step(const unsigned short* __restrict__ hin,
                            unsigned short* __restrict__ hout,
                            float* __restrict__ c,
                            const unsigned short* __restrict__ WhT,
                            const float* __restrict__ xg) {
  __shared__ float lds[4 * 256];
  const int tid  = threadIdx.x;
  const int lane = tid & 31;
  const int g    = tid >> 5;            // wave id == gate id (i,f,g,o)
  const int tm   = blockIdx.x >> 5;     // 0..15 batch tile
  const int tj   = blockIdx.x & 31;     // 0..31 hidden tile
  const int m0   = tm * 16;
  const int j0   = tj * 16;
  const int ln   = lane & 15;
  const int hi16 = lane >> 4;

  // A (16-bit 16x32 layout): lane row = m0+ln; chunks at K+{hi16*8, hi16*8+16}
  const unsigned short* arow = hin + (size_t)(m0 + ln) * H_ + hi16 * 8;
  // B (pre-transposed WhT[n][k]): lane column n = g*H + j0 + ln; chunks at K+{hi16*8, hi16*8+16}
  const unsigned short* brow = WhT + (size_t)(g * H_ + j0 + ln) * H_ + hi16 * 8;

  v8f acc = {};
  // depth-2 software pipeline over 16 K-iterations (K step 32)
  FragAB a0, b0, a1, b1;
  a0.h[0] = *(const v8bf*)(arow);       a0.h[1] = *(const v8bf*)(arow + 16);
  b0.h[0] = *(const v8bf*)(brow);       b0.h[1] = *(const v8bf*)(brow + 16);
  a1.h[0] = *(const v8bf*)(arow + 32);  a1.h[1] = *(const v8bf*)(arow + 48);
  b1.h[0] = *(const v8bf*)(brow + 32);  b1.h[1] = *(const v8bf*)(brow + 48);
  #pragma unroll
  for (int it = 0; it < 14; ++it) {
    const int k = (it + 2) * 32;
    FragAB an, bn;
    an.h[0] = *(const v8bf*)(arow + k);  an.h[1] = *(const v8bf*)(arow + k + 16);
    bn.h[0] = *(const v8bf*)(brow + k);  bn.h[1] = *(const v8bf*)(brow + k + 16);
    acc = WMMA_BF16(a0, b0, acc);
    a0 = a1; b0 = b1; a1 = an; b1 = bn;
  }
  acc = WMMA_BF16(a0, b0, acc);
  acc = WMMA_BF16(a1, b1, acc);

  // Stash this gate's tile: lds[g][m][n] (D layout: lane n=ln, rows hi16*8+v)
  const int mb = hi16 * 8;
  #pragma unroll
  for (int v = 0; v < 8; ++v) lds[g * 256 + (mb + v) * 16 + ln] = acc[v];
  __syncthreads();

  // Fused gate nonlinearities + cell/hidden update: 128 threads x 2 elements
  const float* xgb = xg + j0;
  #pragma unroll
  for (int s = 0; s < 2; ++s) {
    const int e = tid + s * 128;      // 0..255 ; e = m*16 + n
    const int m = e >> 4;
    const int n = e & 15;
    float iv = sigmoidf_(lds[0 * 256 + e] + xgb[0 * H_ + n]);
    float fv = sigmoidf_(lds[1 * 256 + e] + xgb[1 * H_ + n]);
    float gv = tanhf_   (lds[2 * 256 + e] + xgb[2 * H_ + n]);
    float ov = sigmoidf_(lds[3 * 256 + e] + xgb[3 * H_ + n]);
    const size_t idx = (size_t)(m0 + m) * H_ + j0 + n;
    float cn = fv * c[idx] + iv * gv;
    c[idx] = cn;
    hout[idx] = f2bf(ov * tanhf_(cn));
  }
}

// ---------------- Output projection: out[:, t, :] = h @ Wout[t] + bout[t] ----------------
// One block (4 waves) per 16x16 output tile; wave w reduces K in [w*128, w*128+128)
// (4 WMMAs, depth-2 pipelined); partials summed through LDS.
// Grid: 128 blocks -> 512 waves.
__global__ void k_outproj(const unsigned short* __restrict__ h,
                          const unsigned short* __restrict__ WoutT_t,  // (D,H) bf16
                          const float* __restrict__ bout_t,            // (D,)
                          float* __restrict__ out, int t) {
  __shared__ float lds[4 * 256];
  const int tid  = threadIdx.x;
  const int lane = tid & 31;
  const int w    = tid >> 5;            // K-split id 0..3
  const int tm   = blockIdx.x >> 3;     // 0..15
  const int tn   = blockIdx.x & 7;      // 0..7
  const int m0   = tm * 16;
  const int n0   = tn * 16;
  const int ln   = lane & 15;
  const int hi16 = lane >> 4;

  const unsigned short* arow = h + (size_t)(m0 + ln) * H_ + w * 128 + hi16 * 8;
  const unsigned short* brow = WoutT_t + (size_t)(n0 + ln) * H_ + w * 128 + hi16 * 8;

  v8f acc = {};
  FragAB a0, b0, a1, b1;
  a0.h[0] = *(const v8bf*)(arow);       a0.h[1] = *(const v8bf*)(arow + 16);
  b0.h[0] = *(const v8bf*)(brow);       b0.h[1] = *(const v8bf*)(brow + 16);
  a1.h[0] = *(const v8bf*)(arow + 32);  a1.h[1] = *(const v8bf*)(arow + 48);
  b1.h[0] = *(const v8bf*)(brow + 32);  b1.h[1] = *(const v8bf*)(brow + 48);
  #pragma unroll
  for (int it = 0; it < 2; ++it) {
    const int k = (it + 2) * 32;
    FragAB an, bn;
    an.h[0] = *(const v8bf*)(arow + k);  an.h[1] = *(const v8bf*)(arow + k + 16);
    bn.h[0] = *(const v8bf*)(brow + k);  bn.h[1] = *(const v8bf*)(brow + k + 16);
    acc = WMMA_BF16(a0, b0, acc);
    a0 = a1; b0 = b1; a1 = an; b1 = bn;
  }
  acc = WMMA_BF16(a0, b0, acc);
  acc = WMMA_BF16(a1, b1, acc);

  const int mb = hi16 * 8;
  #pragma unroll
  for (int v = 0; v < 8; ++v) lds[w * 256 + (mb + v) * 16 + ln] = acc[v];
  __syncthreads();

  #pragma unroll
  for (int s = 0; s < 2; ++s) {
    const int e = tid + s * 128;      // e = m*16 + n
    const int m = e >> 4;
    const int n = e & 15;
    float sum = lds[e] + lds[256 + e] + lds[512 + e] + lds[768 + e] + bout_t[n0 + n];
    out[(size_t)(m0 + m) * T_ * D_ + (size_t)t * D_ + n0 + n] = sum;
  }
}

// ---------------- launcher ----------------
extern "C" void kernel_launch(void* const* d_in, const int* in_sizes, int n_in,
                              void* d_out, int out_size, void* d_ws, size_t ws_size,
                              hipStream_t stream) {
  const float* z     = (const float*)d_in[0];  // (B,LAT)
  const float* Wz    = (const float*)d_in[1];  // (LAT,H)
  const float* bz    = (const float*)d_in[2];  // (H,)
  const float* token = (const float*)d_in[3];  // (D,)
  const float* Wi    = (const float*)d_in[4];  // (D,4H)
  const float* Wh    = (const float*)d_in[5];  // (H,4H)
  const float* bh    = (const float*)d_in[6];  // (4H,)
  const float* Wout  = (const float*)d_in[7];  // (T,H,D)
  const float* bout  = (const float*)d_in[8];  // (T,D)
  float* out = (float*)d_out;                  // (B,T,D)

  // Workspace carve-up (all 256B aligned by construction)
  char* p = (char*)d_ws;
  unsigned short* WhT   = (unsigned short*)p; p += (size_t)G4H_ * H_ * 2;          // 2 MB
  unsigned short* WoutT = (unsigned short*)p; p += (size_t)T_ * D_ * H_ * 2;       // 32 MB
  float*          xg    = (float*)p;          p += (size_t)G4H_ * 4;               // 8 KB
  float*          c     = (float*)p;          p += (size_t)B_ * H_ * 4;            // 512 KB
  unsigned short* hb0   = (unsigned short*)p; p += (size_t)B_ * H_ * 2;            // 256 KB
  unsigned short* hb1   = (unsigned short*)p;                                      // 256 KB
  (void)in_sizes; (void)n_in; (void)out_size; (void)ws_size;

  // Prep: bf16 conversions/transposes (one-time per call), x_gates+bh, h0/c0.
  k_conv_wh  <<<(G4H_ * H_) / 256, 256, 0, stream>>>(Wh, WhT);
  k_conv_wout<<<(int)(((size_t)T_ * D_ * H_) / 256), 256, 0, stream>>>(Wout, WoutT);
  k_xgates   <<<1, 256, 0, stream>>>(token, Wi, bh, xg);
  k_h0       <<<(B_ * H_) / 256, 256, 0, stream>>>(z, Wz, bz, hb0, c);

  unsigned short* bufs[2] = { hb0, hb1 };
  for (int t = 0; t < T_; ++t) {
    unsigned short* hin  = bufs[t & 1];
    unsigned short* hout = bufs[(t + 1) & 1];
    k_lstm_step<<<512, 128, 0, stream>>>(hin, hout, c, WhT, xg);
    k_outproj  <<<128, 128, 0, stream>>>(hout, WoutT + (size_t)t * D_ * H_,
                                         bout + (size_t)t * D_, out, t);
  }
}
